// SelfAttention_52707838657226
// MI455X (gfx1250) — matrix operand order repaired and checked by reading, not compiled, not run on hardware
//
#include <hip/hip_runtime.h>

typedef __attribute__((ext_vector_type(16))) __bf16        v16bf;
typedef __attribute__((ext_vector_type(8)))  float         v8f;
typedef __attribute__((ext_vector_type(8)))  unsigned int  v8u;

// ---------- helpers ----------

__device__ __forceinline__ unsigned short f2bf(float f) {
    union { float f; unsigned int u; } x; x.f = f;
    unsigned int u = x.u;
    return (unsigned short)((u + 0x7FFFu + ((u >> 16) & 1u)) >> 16);   // RNE
}
__device__ __forceinline__ unsigned int pack2bf(float a, float b) {
    return (unsigned int)f2bf(a) | ((unsigned int)f2bf(b) << 16);
}

__device__ __forceinline__ v8f wmma_bf16(v8u a, v8u b, v8f c) {
    return __builtin_amdgcn_wmma_f32_16x16x32_bf16(
        false, __builtin_bit_cast(v16bf, a),
        false, __builtin_bit_cast(v16bf, b),
        (short)0, c, false, false);
}

// A-fragment (16x32, bf16) from a row-major [16 rows x K] matrix.
// Lane L holds row M=L%16. dwords: kd + half*8 + {0..7}.
__device__ __forceinline__ v8u frag_a(const unsigned int* row, int kd, int half) {
    const uint4* p = (const uint4*)(row + kd + half * 8);
    uint4 lo = p[0];
    uint4 hi = p[1];
    v8u r = {lo.x, lo.y, lo.z, lo.w, hi.x, hi.y, hi.z, hi.w};
    return r;
}
// B-fragment (32x16, bf16) of B[K][N] sourced from row-major R[N][K] (R[n] = B[:,n]).
// Lane L holds col N=L%16. dwords: kd + half*8 + {0..7}.
__device__ __forceinline__ v8u frag_b(const unsigned int* row, int kd, int half) {
    const uint4* p = (const uint4*)(row + kd + half * 8);
    uint4 lo = p[0];
    uint4 hi = p[1];
    v8u r = {lo.x, lo.y, lo.z, lo.w, hi.x, hi.y, hi.z, hi.w};
    return r;
}

__device__ __forceinline__ float swz16f(float v) {   // exchange with lane^16 (SWAPX16)
    int i = __builtin_amdgcn_ds_swizzle(__builtin_bit_cast(int, v), 0x401F);
    return __builtin_bit_cast(float, i);
}
__device__ __forceinline__ unsigned int swz16u(unsigned int v) {
    return (unsigned int)__builtin_amdgcn_ds_swizzle((int)v, 0x401F);
}

// ---------- kernel 1: weight f32 -> bf16 ----------

__global__ void wconv_kernel(const float* __restrict__ wq, const float* __restrict__ wk,
                             const float* __restrict__ wv,
                             unsigned short* __restrict__ wqb, unsigned short* __restrict__ wkb,
                             unsigned short* __restrict__ wvb) {
    int i = blockIdx.x * blockDim.x + threadIdx.x;
    if (i < 32768) { wqb[i] = f2bf(wq[i]); wkb[i] = f2bf(wk[i]); }
    if (i < 262144) { wvb[i] = f2bf(wv[i]); }
}

// ---------- kernel 2: projections  Q[B,T,64], K[B,T,64], VT[B,512,T]  (bf16) ----------
// One block per (b, 16-t tile). x tile [512 c' x 16 t] packed into LDS as
// xb[t][c2] = pack(x[2c2][t], x[2c2+1][t]); this is the B-fragment source (lane = t).

__global__ __launch_bounds__(128) void proj_kernel(
    const float* __restrict__ x,
    const unsigned int* __restrict__ wqb,   // [64 ][256] dwords
    const unsigned int* __restrict__ wkb,   // [64 ][256]
    const unsigned int* __restrict__ wvb,   // [512][256]
    const float* __restrict__ bq, const float* __restrict__ bk, const float* __restrict__ bv,
    unsigned int* __restrict__ Qb,          // [B,2048,32] dwords
    unsigned int* __restrict__ Kb,          // [B,2048,32]
    unsigned int* __restrict__ VTb)         // [B,512,1024]
{
    __shared__ unsigned int xb[16 * 256];   // 16 KB
    const int blk = blockIdx.x;
    const int b   = blk >> 7;
    const int t0  = (blk & 127) << 4;
    const int tid = threadIdx.x;

    const float* x0 = x + (size_t)b * 512 * 2048;
    for (int item = tid; item < 1024; item += 128) {
        int c2 = item >> 2;
        int tq = item & 3;
        float4 a = *(const float4*)(x0 + (size_t)(2 * c2)     * 2048 + t0 + 4 * tq);
        float4 c = *(const float4*)(x0 + (size_t)(2 * c2 + 1) * 2048 + t0 + 4 * tq);
        xb[(4 * tq + 0) * 256 + c2] = pack2bf(a.x, c.x);
        xb[(4 * tq + 1) * 256 + c2] = pack2bf(a.y, c.y);
        xb[(4 * tq + 2) * 256 + c2] = pack2bf(a.z, c.z);
        xb[(4 * tq + 3) * 256 + c2] = pack2bf(a.w, c.w);
    }
    __syncthreads();

    const int wave = tid >> 5, lane = tid & 31;
    const int half = lane >> 4, ln = lane & 15;
    const unsigned int* lrow = xb + ln * 256;

    for (int it = 0; it < 10; ++it) {
        int id = wave + 4 * it;             // 0..3 Q, 4..7 K, 8..39 VT
        int kind, m0;
        const unsigned int* wrow;
        if (id < 4)      { kind = 0; m0 = id * 16;        wrow = wqb + (size_t)(m0 + ln) * 256; }
        else if (id < 8) { kind = 1; m0 = (id - 4) * 16;  wrow = wkb + (size_t)(m0 + ln) * 256; }
        else             { kind = 2; m0 = (id - 8) * 16;  wrow = wvb + (size_t)(m0 + ln) * 256; }

        v8f acc = {0.f, 0.f, 0.f, 0.f, 0.f, 0.f, 0.f, 0.f};
        #pragma unroll
        for (int k = 0; k < 16; ++k) {
            acc = wmma_bf16(frag_a(wrow, k * 16, half), frag_b(lrow, k * 16, half), acc);
        }
        const float* bias = (kind == 0) ? bq : (kind == 1) ? bk : bv;
        #pragma unroll
        for (int r = 0; r < 8; ++r) acc[r] += bias[m0 + r + half * 8];

        if (kind < 2) {                     // store D[o][t] to [t][o] row-major bf16
            unsigned int* orow = ((kind == 0) ? Qb : Kb) + (size_t)(b * 2048 + t0 + ln) * 32;
            int od = (m0 >> 1) + half * 4;
            #pragma unroll
            for (int i = 0; i < 4; ++i) orow[od + i] = pack2bf(acc[2 * i], acc[2 * i + 1]);
        } else {                            // VT[c][t] element stores
            unsigned short* vt = (unsigned short*)VTb;
            #pragma unroll
            for (int r = 0; r < 8; ++r) {
                int c = m0 + r + half * 8;
                vt[(size_t)(b * 512 + c) * 2048 + t0 + ln] = f2bf(acc[r]);
            }
        }
    }
}

// ---------- kernel 3: flash attention + residual ----------
// One block per (b, 32-t tile); 8 waves = 2 t-subtiles x 4 channel quarters.
// Wave handles t-subtile (wave>>2) and channels [(wave&3)*128, +128).
// Sᵀ = K·Qᵀ (lane = t, VGPR = s) -> online softmax per lane -> Pᵀ B-frag via
// one SWAPX16 exchange -> Oᵀ += Vᵀ·Pᵀ.  Halves per-batch L2 traffic on Vᵀ
// versus a 16-t block, and ballot-skips the accumulator rescale once the
// running max has stabilized (alpha == 1 exactly).

__global__ __launch_bounds__(256) void attn_kernel(
    const unsigned int* __restrict__ Qb,    // [B,2048,32] dwords
    const unsigned int* __restrict__ Kb,    // [B,2048,32]
    const unsigned int* __restrict__ VTb,   // [B,512,1024]
    const float* __restrict__ x,
    const float* __restrict__ gamma,
    float* __restrict__ out)
{
    const int blk = blockIdx.x;
    const int b   = blk >> 6;               // 64 blocks of 32 t's per batch
    const int tb  = (blk & 63) << 5;
    const int tid = threadIdx.x;
    const int wave = tid >> 5, lane = tid & 31;
    const int half = lane >> 4, ln = lane & 15;
    const int t0    = tb + ((wave >> 2) << 4);
    const int cbase = (wave & 3) * 128;

    // Q B-fragments (Qᵀ), o-chunks [0,32) and [32,64)
    const unsigned int* qrow = Qb + (size_t)(b * 2048 + t0 + ln) * 32;
    v8u qf0 = frag_b(qrow, 0, half);
    v8u qf1 = frag_b(qrow, 16, half);

    v8f acc[8];
    #pragma unroll
    for (int i = 0; i < 8; ++i) {
        #pragma unroll
        for (int j = 0; j < 8; ++j) acc[i][j] = 0.f;
    }
    float m = -3.0e38f, l = 0.f;

    for (int s0 = 0; s0 < 2048; s0 += 32) {
        // ---- scores: Sᵀ tiles, sub = 0,1 (s = s0+16*sub + ...) ----
        v8f st[2];
        #pragma unroll
        for (int sub = 0; sub < 2; ++sub) {
            const unsigned int* krow = Kb + (size_t)(b * 2048 + s0 + sub * 16 + ln) * 32;
            v8f c0 = {0.f, 0.f, 0.f, 0.f, 0.f, 0.f, 0.f, 0.f};
            c0 = wmma_bf16(frag_a(krow, 0, half), qf0, c0);
            c0 = wmma_bf16(frag_a(krow, 16, half), qf1, c0);
            st[sub] = c0;
        }
        // ---- online softmax over this 32-key chunk (per lane: fixed t) ----
        float cm = st[0][0];
        #pragma unroll
        for (int r = 1; r < 8; ++r) cm = fmaxf(cm, st[0][r]);
        #pragma unroll
        for (int r = 0; r < 8; ++r) cm = fmaxf(cm, st[1][r]);
        cm = fmaxf(cm, swz16f(cm));                  // combine with partner half
        float mnew = fmaxf(m, cm);
        // Wave-uniform skip: if no lane raised its max, alpha == 1 exactly.
        if (__builtin_amdgcn_ballot_w32(mnew > m) != 0u) {
            float alpha = __expf(m - mnew);
            l *= alpha;
            #pragma unroll
            for (int i = 0; i < 8; ++i) {
                #pragma unroll
                for (int j = 0; j < 8; ++j) acc[i][j] *= alpha;
            }
            m = mnew;
        }
        float e0[8], e1[8], psum = 0.f;
        #pragma unroll
        for (int r = 0; r < 8; ++r) { e0[r] = __expf(st[0][r] - m); psum += e0[r]; }
        #pragma unroll
        for (int r = 0; r < 8; ++r) { e1[r] = __expf(st[1][r] - m); psum += e1[r]; }
        psum += swz16f(psum);
        l += psum;
        // ---- build Pᵀ B-fragment (K-dim = s within chunk) ----
        unsigned int p0[4], p1[4], pp0[4], pp1[4];
        #pragma unroll
        for (int i = 0; i < 4; ++i) {
            p0[i] = pack2bf(e0[2 * i], e0[2 * i + 1]);
            p1[i] = pack2bf(e1[2 * i], e1[2 * i + 1]);
        }
        #pragma unroll
        for (int i = 0; i < 4; ++i) { pp0[i] = swz16u(p0[i]); pp1[i] = swz16u(p1[i]); }
        v8u pf;
        #pragma unroll
        for (int i = 0; i < 4; ++i) {
            pf[i]     = half ? pp1[i] : p0[i];       // K = {0..7 | 16..23}
            pf[4 + i] = half ? p1[i]  : pp0[i];      // K = {8..15 | 24..31}
        }
        // ---- Oᵀ += Vᵀ · Pᵀ over this wave's 128 channels ----
        #pragma unroll
        for (int ct = 0; ct < 8; ++ct) {
            const unsigned int* vrow =
                VTb + (size_t)(b * 512 + cbase + ct * 16 + ln) * 1024;
            acc[ct] = wmma_bf16(frag_a(vrow, s0 >> 1, half), pf, acc[ct]);
        }
    }

    // ---- normalize, gamma * O + x  (fold gamma/l into one per-lane scale) ----
    const float gs = gamma[0] / l;
    #pragma unroll
    for (int ct = 0; ct < 8; ++ct) {
        #pragma unroll
        for (int r = 0; r < 8; ++r) {
            int c = cbase + ct * 16 + r + half * 8;
            size_t idx = (size_t)(b * 512 + c) * 2048 + t0 + ln;
            out[idx] = gs * acc[ct][r] + x[idx];
        }
    }
}

// ---------- launch ----------

extern "C" void kernel_launch(void* const* d_in, const int* in_sizes, int n_in,
                              void* d_out, int out_size, void* d_ws, size_t ws_size,
                              hipStream_t stream) {
    (void)in_sizes; (void)n_in; (void)out_size; (void)ws_size;
    const float* x     = (const float*)d_in[0];
    const float* wq    = (const float*)d_in[1];
    const float* bq    = (const float*)d_in[2];
    const float* wk    = (const float*)d_in[3];
    const float* bk    = (const float*)d_in[4];
    const float* wv    = (const float*)d_in[5];
    const float* bv    = (const float*)d_in[6];
    const float* gamma = (const float*)d_in[7];

    char* ws = (char*)d_ws;
    unsigned short* wqb = (unsigned short*)(ws + 0);        //  64*512*2 = 64 KB
    unsigned short* wkb = (unsigned short*)(ws + 65536);    //  64 KB
    unsigned short* wvb = (unsigned short*)(ws + 131072);   // 512*512*2 = 512 KB
    unsigned int*   Qb  = (unsigned int*)(ws + 655360);     // 8*2048*64*2 = 2 MB
    unsigned int*   Kb  = (unsigned int*)(ws + 2752512);    // 2 MB
    unsigned int*   VTb = (unsigned int*)(ws + 4849664);    // 8*512*2048*2 = 16 MB

    wconv_kernel<<<dim3(1024), dim3(256), 0, stream>>>(wq, wk, wv, wqb, wkb, wvb);
    proj_kernel<<<dim3(1024), dim3(128), 0, stream>>>(
        x, (const unsigned int*)wqb, (const unsigned int*)wkb, (const unsigned int*)wvb,
        bq, bk, bv, Qb, Kb, VTb);
    attn_kernel<<<dim3(512), dim3(256), 0, stream>>>(
        (const unsigned int*)Qb, (const unsigned int*)Kb, (const unsigned int*)VTb,
        x, gamma, (float*)d_out);
}
